// GRU_52364241272976
// MI455X (gfx1250) — matrix operand order repaired and checked by reading, not compiled, not run on hardware
//
#include <hip/hip_runtime.h>
#include <hip/hip_bf16.h>

#define HIDDEN   1024
#define BATCH    64
#define TSTEPS   512
#define INSZ     512
#define NWG      32
#define TPB      256
#define NWAVES   (NWG * TPB / 32)   // 256 waves
#define NTHREADS (NWG * TPB)        // 8192 threads

#define KC       128                // K-chunk staged in LDS
#define APITCH   (KC + 8)           // halves; +16B row pad to spread LDS banks

typedef __attribute__((ext_vector_type(16))) _Float16 v16h;
typedef __attribute__((ext_vector_type(8)))  _Float16 v8h;
typedef __attribute__((ext_vector_type(8)))  float    v8f;

// ---------------------------------------------------------------------------
// helpers
// ---------------------------------------------------------------------------
__device__ __forceinline__ float sigmoidf_(float x) {
    return 1.0f / (1.0f + __expf(-x));
}

__device__ __forceinline__ v8h cvt8(float4 lo, float4 hi) {
    v8h o = { (_Float16)lo.x, (_Float16)lo.y, (_Float16)lo.z, (_Float16)lo.w,
              (_Float16)hi.x, (_Float16)hi.y, (_Float16)hi.z, (_Float16)hi.w };
    return o;
}

// Device-wide barrier: monotonically increasing counter, one arrive per WG.
__device__ __forceinline__ void grid_sync(unsigned* cnt, unsigned target) {
    __builtin_amdgcn_fence(__ATOMIC_RELEASE, "agent");
    __syncthreads();
    if (threadIdx.x == 0) {
        __hip_atomic_fetch_add(cnt, 1u, __ATOMIC_RELEASE, __HIP_MEMORY_SCOPE_AGENT);
        while (__hip_atomic_load(cnt, __ATOMIC_ACQUIRE, __HIP_MEMORY_SCOPE_AGENT) < target)
            __builtin_amdgcn_s_sleep(1);
    }
    __syncthreads();
    __builtin_amdgcn_fence(__ATOMIC_ACQUIRE, "agent");
}

// ---------------------------------------------------------------------------
// elementwise build phases (all threads) — vectorized: 2x b128 loads + one
// 16B f16 store per iteration. These phases sit between device barriers on
// the sequential critical path, so instruction count matters directly.
// ---------------------------------------------------------------------------
__device__ __forceinline__ void conv_w(_Float16* dst, const float* src, int n, int tid) {
    int n8 = n >> 3;
    for (int i = tid; i < n8; i += NTHREADS) {
        const float4* s = (const float4*)(src + (size_t)i * 8);
        *(v8h*)(dst + (size_t)i * 8) = cvt8(s[0], s[1]);
    }
}

// ih[:, colOff:colOff+ncols] = (f16) src (row stride srcStride); ncols = 1<<lg
__device__ __forceinline__ void build_cast(_Float16* ih, int K, int colOff,
                                           const float* src, int srcStride,
                                           int lg, int tid) {
    int lg8 = lg - 3;                       // 8-element groups per row
    int total8 = BATCH << lg8;
    for (int i = tid; i < total8; i += NTHREADS) {
        int row = i >> lg8;
        int col = (i & ((1 << lg8) - 1)) << 3;
        const float4* s = (const float4*)(src + (size_t)row * srcStride + col);
        *(v8h*)(ih + (size_t)row * K + colOff + col) = cvt8(s[0], s[1]);
    }
}

// ih[:, colOff:colOff+H] = (f16)(r * h)   (r,h contiguous [64][1024])
__device__ __forceinline__ void build_rh(_Float16* ih, int K, int colOff,
                                         const float* rbuf, const float* h, int tid) {
    const int total8 = (BATCH * HIDDEN) >> 3;
    for (int i = tid; i < total8; i += NTHREADS) {
        int row = i >> 7;                   // 128 groups of 8 per row
        int col = (i & 127) << 3;
        const float4* rp = (const float4*)(rbuf + (size_t)i * 8);
        const float4* hp = (const float4*)(h + (size_t)i * 8);
        float4 r0 = rp[0], r1 = rp[1], h0 = hp[0], h1 = hp[1];
        float4 lo = make_float4(r0.x * h0.x, r0.y * h0.y, r0.z * h0.z, r0.w * h0.w);
        float4 hi = make_float4(r1.x * h1.x, r1.y * h1.y, r1.z * h1.z, r1.w * h1.w);
        *(v8h*)(ih + (size_t)row * K + colOff + col) = cvt8(lo, hi);
    }
}

// ---------------------------------------------------------------------------
// Async (fire-and-forget) staging of one A chunk: 64 rows x KC halves of ih
// copied global->LDS via GLOBAL_LOAD_ASYNC_TO_LDS_B128 (ASYNCcnt path; no
// VGPR round-trip, no loadcnt stall). 1024 16B pieces, exactly 4 per thread
// (fixed trip count -> no EXEC predication around the async issues).
// LDS destination address = low 32 bits of the generic pointer (flat LDS
// addressing truncates to addr[31:0]).
// ---------------------------------------------------------------------------
__device__ __forceinline__ void stage_chunk_async(_Float16* dst, const _Float16* ih,
                                                  int K, int k0, int tl) {
    #pragma unroll
    for (int it = 0; it < (64 * (KC / 8)) / TPB; ++it) {   // 4 iterations
        int c   = tl + it * TPB;
        int row = c >> 4;       // KC/8 == 16 chunks per row
        int cc  = c & 15;
        unsigned ldsOff = (unsigned)(uintptr_t)(dst + row * APITCH + cc * 8);
        unsigned long long ga =
            (unsigned long long)(ih + (size_t)row * K + k0 + cc * 8);
        asm volatile("global_load_async_to_lds_b128 %0, %1, off"
                     :: "v"(ldsOff), "v"(ga)
                     : "memory");
    }
}

__device__ __forceinline__ void wait_async0() {
    asm volatile("s_wait_asynccnt 0" ::: "memory");
}

__device__ __forceinline__ v16h a_frag_lds(const _Float16* A, int base, int kk, int aHalf) {
    union { v16h v; v8h h[2]; } u;
    u.h[0] = *(const v8h*)(A + base + kk + aHalf);
    u.h[1] = *(const v8h*)(A + base + kk + 16 + aHalf);
    return u.v;
}

// ---------------------------------------------------------------------------
// Core GEMM for one task (16-col strip s, 32-row half mh) with A double-
// buffered through LDS via async copies; B (weight rows) streamed from
// global (L2-resident). Next chunk's async copy is issued BEFORE this
// chunk's WMMAs; its s_wait_asynccnt lands after the math, so staging is
// fully hidden. Must be called by ALL threads of an active workgroup.
// ---------------------------------------------------------------------------
__device__ __forceinline__ void gemm_tiles(const _Float16* ih, const _Float16* W,
                                           int K, int task, int lane, int tl,
                                           _Float16 (*abuf)[64 * APITCH],
                                           v8f& acc0, v8f& acc1)
{
    int s = task >> 1, mh = task & 1;
    int n = s * 16 + (lane & 15);
    const _Float16* wrow = W + (size_t)n * K + ((lane >> 4) << 3);
    int aHalf = (lane >> 4) << 3;
    const int ar0 = (mh * 32 + (lane & 15)) * APITCH;
    const int ar1 = ar0 + 16 * APITCH;

    acc0 = (v8f){}; acc1 = (v8f){};
    const int nChunks = K / KC;

    stage_chunk_async(abuf[0], ih, K, 0, tl);
    wait_async0();
    __syncthreads();
    for (int c = 0; c < nChunks; ++c) {
        if (c + 1 < nChunks) stage_chunk_async(abuf[(c + 1) & 1], ih, K, (c + 1) * KC, tl);
        const _Float16* A = abuf[c & 1];
        const int kbase = c * KC;
        #pragma unroll
        for (int kk = 0; kk < KC; kk += 32) {
            v16h b  = __builtin_shufflevector(*(const v8h*)(wrow + kbase + kk), *(const v8h*)(wrow + kbase + kk + 16), 0, 1, 2, 3, 4, 5, 6, 7, 8, 9, 10, 11, 12, 13, 14, 15);
            v16h a0 = a_frag_lds(A, ar0, kk, aHalf);
            v16h a1 = a_frag_lds(A, ar1, kk, aHalf);
            acc0 = __builtin_amdgcn_wmma_f32_16x16x32_f16(false, a0, false, b, (short)0, acc0, false, false);
            acc1 = __builtin_amdgcn_wmma_f32_16x16x32_f16(false, a1, false, b, (short)0, acc1, false, false);
        }
        wait_async0();          // staged copies landed (hidden behind WMMAs)
        __syncthreads();
    }
}

// ---------------------------------------------------------------------------
// GEMM phases. One task per wave; task counts are multiples of 8 so whole
// workgroups are uniformly active/idle (syncthreads-safe, EXEC all-ones).
// ---------------------------------------------------------------------------
__device__ __forceinline__ void gemm_rz(const _Float16* ih, const _Float16* W,
                                        const float* bias, float* rbuf, float* zbuf,
                                        int K, int wave, int lane, int tl,
                                        _Float16 (*abuf)[64 * APITCH]) {
    const int nTasks = (2 * HIDDEN / 16) * 2;   // 256
    if (wave >= nTasks) return;                  // WG-uniform
    v8f acc0, acc1;
    gemm_tiles(ih, W, K, wave, lane, tl, abuf, acc0, acc1);

    int n  = (wave >> 1) * 16 + (lane & 15);
    int mh = wave & 1;
    float bv = bias[n];
    int rbase = mh * 32 + ((lane >> 4) << 3);
    float* dst; int col;
    if (n < HIDDEN) { dst = rbuf; col = n; } else { dst = zbuf; col = n - HIDDEN; }
    #pragma unroll
    for (int j = 0; j < 8; ++j) {
        dst[(size_t)(rbase + j)      * HIDDEN + col] = sigmoidf_(acc0[j] + bv);
        dst[(size_t)(rbase + 16 + j) * HIDDEN + col] = sigmoidf_(acc1[j] + bv);
    }
}

__device__ __forceinline__ void gemm_n(const _Float16* ih, const _Float16* W,
                                       const float* bias, const float* zbuf, float* h,
                                       int K, int wave, int lane, int tl,
                                       _Float16 (*abuf)[64 * APITCH]) {
    const int nTasks = (HIDDEN / 16) * 2;       // 128
    if (wave >= nTasks) return;                  // WG-uniform
    v8f acc0, acc1;
    gemm_tiles(ih, W, K, wave, lane, tl, abuf, acc0, acc1);

    int n  = (wave >> 1) * 16 + (lane & 15);
    int mh = wave & 1;
    float bv = bias[n];
    int rbase = mh * 32 + ((lane >> 4) << 3);
    #pragma unroll
    for (int j = 0; j < 8; ++j) {
        size_t i0 = (size_t)(rbase + j)      * HIDDEN + n;
        size_t i1 = (size_t)(rbase + 16 + j) * HIDDEN + n;
        float n0 = tanhf(acc0[j] + bv), n1 = tanhf(acc1[j] + bv);
        float z0 = zbuf[i0], z1 = zbuf[i1];
        h[i0] = (1.0f - z0) * n0 + z0 * h[i0];
        h[i1] = (1.0f - z1) * n1 + z1 * h[i1];
    }
}

__device__ __forceinline__ void gemm_fc(const _Float16* ih, const _Float16* W,
                                        const float* bias, float* out,
                                        int K, int wave, int lane, int tl,
                                        _Float16 (*abuf)[64 * APITCH]) {
    const int nTasks = (INSZ / 16) * 2;          // 64
    if (wave >= nTasks) return;                  // WG-uniform
    v8f acc0, acc1;
    gemm_tiles(ih, W, K, wave, lane, tl, abuf, acc0, acc1);

    int n  = (wave >> 1) * 16 + (lane & 15);
    int mh = wave & 1;
    float bv = bias[n];
    int rbase = mh * 32 + ((lane >> 4) << 3);
    #pragma unroll
    for (int j = 0; j < 8; ++j) {
        out[(size_t)(rbase + j)      * INSZ + n] = acc0[j] + bv;
        out[(size_t)(rbase + 16 + j) * INSZ + n] = acc1[j] + bv;
    }
}

// ---------------------------------------------------------------------------
// persistent GRU kernel: init -> 512 x (8 phases w/ device barriers) -> FC
// ---------------------------------------------------------------------------
__global__ void __launch_bounds__(TPB, 1)
gru_persistent(const float* __restrict__ x,
               const float* __restrict__ Wrz0, const float* __restrict__ brz0,
               const float* __restrict__ Wn0,  const float* __restrict__ bn0,
               const float* __restrict__ Wrz1, const float* __restrict__ brz1,
               const float* __restrict__ Wn1,  const float* __restrict__ bn1,
               const float* __restrict__ Wfc,  const float* __restrict__ bfc,
               float* __restrict__ out,
               unsigned* cnt,
               float* h0, float* h1, float* rbuf, float* zbuf,
               _Float16* ih,
               _Float16* Wrz0h, _Float16* Wn0h, _Float16* Wrz1h, _Float16* Wn1h,
               _Float16* Wfch)
{
    __shared__ _Float16 abuf[2][64 * APITCH];   // ~34 KB of the 320 KB/WGP pool

    const int tid  = blockIdx.x * TPB + threadIdx.x;
    const int wave = tid >> 5;
    const int lane = threadIdx.x & 31;
    const int tl   = threadIdx.x;
    unsigned target = 0;

    // ---- init: f32 -> f16 weights (L2-resident thereafter), zero states ----
    conv_w(Wrz0h, Wrz0, 2 * HIDDEN * (INSZ + HIDDEN), tid);
    conv_w(Wn0h,  Wn0,      HIDDEN * (INSZ + HIDDEN), tid);
    conv_w(Wrz1h, Wrz1, 2 * HIDDEN * (2 * HIDDEN),    tid);
    conv_w(Wn1h,  Wn1,      HIDDEN * (2 * HIDDEN),    tid);
    conv_w(Wfch,  Wfc,      INSZ * HIDDEN,            tid);
    for (int i = tid; i < BATCH * HIDDEN; i += NTHREADS) { h0[i] = 0.0f; h1[i] = 0.0f; }
    target += NWG; grid_sync(cnt, target);

    const int K0 = INSZ + HIDDEN;   // 1536
    const int K1 = 2 * HIDDEN;      // 2048

    for (int t = 0; t < TSTEPS; ++t) {
        // ---------------- layer 0 ----------------
        build_cast(ih, K0, 0,    x + (size_t)t * INSZ, TSTEPS * INSZ, 9, tid); // xt
        build_cast(ih, K0, INSZ, h0, HIDDEN, 10, tid);                          // h0
        target += NWG; grid_sync(cnt, target);

        gemm_rz(ih, Wrz0h, brz0, rbuf, zbuf, K0, wave, lane, tl, abuf);
        target += NWG; grid_sync(cnt, target);

        build_rh(ih, K0, INSZ, rbuf, h0, tid);                                  // r*h0
        target += NWG; grid_sync(cnt, target);

        gemm_n(ih, Wn0h, bn0, zbuf, h0, K0, wave, lane, tl, abuf);              // h0'
        target += NWG; grid_sync(cnt, target);

        // ---------------- layer 1 ----------------
        build_cast(ih, K1, 0,      h0, HIDDEN, 10, tid);
        build_cast(ih, K1, HIDDEN, h1, HIDDEN, 10, tid);
        target += NWG; grid_sync(cnt, target);

        gemm_rz(ih, Wrz1h, brz1, rbuf, zbuf, K1, wave, lane, tl, abuf);
        target += NWG; grid_sync(cnt, target);

        build_rh(ih, K1, HIDDEN, rbuf, h1, tid);                                // r*h1
        target += NWG; grid_sync(cnt, target);

        gemm_n(ih, Wn1h, bn1, zbuf, h1, K1, wave, lane, tl, abuf);              // h1'
        target += NWG; grid_sync(cnt, target);
    }

    // ---- final fc: out = h1 @ Wfc.T + bfc ----
    build_cast(ih, HIDDEN, 0, h1, HIDDEN, 10, tid);
    target += NWG; grid_sync(cnt, target);
    gemm_fc(ih, Wfch, bfc, out, HIDDEN, wave, lane, tl, abuf);
}

// ---------------------------------------------------------------------------
extern "C" void kernel_launch(void* const* d_in, const int* in_sizes, int n_in,
                              void* d_out, int out_size, void* d_ws, size_t ws_size,
                              hipStream_t stream) {
    (void)in_sizes; (void)n_in; (void)out_size; (void)ws_size;

    const float* x    = (const float*)d_in[0];
    const float* Wrz0 = (const float*)d_in[1];
    const float* brz0 = (const float*)d_in[2];
    const float* Wn0  = (const float*)d_in[3];
    const float* bn0  = (const float*)d_in[4];
    const float* Wrz1 = (const float*)d_in[5];
    const float* brz1 = (const float*)d_in[6];
    const float* Wn1  = (const float*)d_in[7];
    const float* bn1  = (const float*)d_in[8];
    const float* Wfc  = (const float*)d_in[9];
    const float* bfc  = (const float*)d_in[10];
    float* out = (float*)d_out;

    char*  ws  = (char*)d_ws;
    size_t off = 0;
    auto alloc = [&](size_t bytes) -> void* {
        void* p = ws + off;
        off = (off + bytes + 255) & ~(size_t)255;
        return p;
    };

    unsigned* cnt  = (unsigned*)alloc(256);
    float* h0      = (float*)alloc((size_t)BATCH * HIDDEN * 4);
    float* h1      = (float*)alloc((size_t)BATCH * HIDDEN * 4);
    float* rbuf    = (float*)alloc((size_t)BATCH * HIDDEN * 4);
    float* zbuf    = (float*)alloc((size_t)BATCH * HIDDEN * 4);
    _Float16* ih    = (_Float16*)alloc((size_t)BATCH * 2 * HIDDEN * 2);
    _Float16* Wrz0h = (_Float16*)alloc((size_t)2 * HIDDEN * (INSZ + HIDDEN) * 2);
    _Float16* Wn0h  = (_Float16*)alloc((size_t)HIDDEN * (INSZ + HIDDEN) * 2);
    _Float16* Wrz1h = (_Float16*)alloc((size_t)2 * HIDDEN * 2 * HIDDEN * 2);
    _Float16* Wn1h  = (_Float16*)alloc((size_t)HIDDEN * 2 * HIDDEN * 2);
    _Float16* Wfch  = (_Float16*)alloc((size_t)INSZ * HIDDEN * 2);

    hipMemsetAsync(cnt, 0, 256, stream);

    gru_persistent<<<NWG, TPB, 0, stream>>>(
        x, Wrz0, brz0, Wn0, bn0, Wrz1, brz1, Wn1, bn1, Wfc, bfc,
        out, cnt, h0, h1, rbuf, zbuf, ih,
        Wrz0h, Wn0h, Wrz1h, Wn1h, Wfch);
}
